// GraphTrajectoryLSTM_16913581211666
// MI455X (gfx1250) — matrix operand order repaired and checked by reading, not compiled, not run on hardware
//
#include <hip/hip_runtime.h>
#include <hip/hip_bf16.h>
#include <math.h>

typedef _Float16 half_t;
typedef __attribute__((ext_vector_type(16))) _Float16 v16h;
typedef __attribute__((ext_vector_type(8)))  _Float16 h8;
typedef __attribute__((ext_vector_type(8)))  float    v8f;

// ---------------------------------------------------------------------------
// WMMA f16 (16x16x32, f32 accum) fragment index helpers (contiguous K order, same for A and B)
// A (16x32 f16): lane holds row M=lane%16; element j -> K = (lane/16)*16 + j
//   => per lane: 16 contiguous K at (lane/16)*16
// B (32x16 f16): lane holds col N=lane%16; element j -> K = (lane/16)*16 + j
//   => per lane: 16 contiguous K at (lane/16)*16
// C/D (16x16 f32): element v -> M = v + 8*(lane/16), N = lane%16
// ---------------------------------------------------------------------------
__device__ __forceinline__ int a_k_of(int lane, int j) {
    return ((lane >> 4) << 4) + j;
}
__device__ __forceinline__ int b_k_of(int lane, int j) {
    return ((lane >> 4) << 4) + j;
}

__device__ __forceinline__ void put8f(v16h& f, int base, float4 a, float4 b) {
    f[base + 0] = (half_t)a.x; f[base + 1] = (half_t)a.y;
    f[base + 2] = (half_t)a.z; f[base + 3] = (half_t)a.w;
    f[base + 4] = (half_t)b.x; f[base + 5] = (half_t)b.y;
    f[base + 6] = (half_t)b.z; f[base + 7] = (half_t)b.w;
}
__device__ __forceinline__ void put8h(v16h& f, int base, h8 v) {
#pragma unroll
    for (int i = 0; i < 8; ++i) f[base + i] = v[i];
}

__device__ __forceinline__ float sigf(float x)      { return 1.f / (1.f + __expf(-x)); }
__device__ __forceinline__ float softplusf(float x) { return (x > 20.f) ? x : log1pf(__expf(x)); }

// ---------------------------------------------------------------------------
// Generic batched GEMM: C[b] = act( A[b] @ op(B[b]) + bias1[b] + bias2[b] )
// f32 in/out, f16 WMMA f32-accumulate. 256 threads = 8 waves; wave -> one
// 16x32 strip (two 16x16 N-tiles sharing one A-fragment: 2x arithmetic
// intensity). Macro tile 64(M) x 64(N). Interior tiles: unconditional
// vectorized loads (float4 -> global_load_b128). Edge tiles / K tails:
// branchless clamped loads + selects (no EXEC-mask churn).
// ---------------------------------------------------------------------------
__global__ void wmma_gemm_kernel(
    const float* __restrict__ A, int lda, long sA,
    const float* __restrict__ B, int ldb, long sB, int transB,
    const float* __restrict__ bias1, const float* __restrict__ bias2, long sBias,
    float* __restrict__ C, int ldc, long sC,
    int M, int N, int K, int act)
{
    const int batch = blockIdx.z;
    const float* Ab = A + (long)batch * sA;
    const float* Bb = B + (long)batch * sB;
    float*       Cb = C + (long)batch * sC;

    const int lane = threadIdx.x & 31;
    const int wave = threadIdx.x >> 5;
    const int m0 = blockIdx.x * 64 + (wave & 3) * 16;
    const int n0 = blockIdx.y * 64 + (wave >> 2) * 32;
    if (m0 >= M || n0 >= N) return;          // wave-uniform: EXEC stays full

    const int  row    = m0 + (lane & 15);
    const int  col0   = n0 + (lane & 15);
    const int  col1   = col0 + 16;
    const bool rowOK  = row  < M;
    const bool col0OK = col0 < N;
    const bool col1OK = col1 < N;
    const int  rc  = rowOK  ? row  : (M - 1);   // clamped (always safe to load)
    const int  cc0 = col0OK ? col0 : (N - 1);
    const int  cc1 = col1OK ? col1 : (N - 1);

    const float* Arow = Ab + (long)rc * lda;
    const float* Bt0  = Bb + (long)cc0 * ldb;   // transB rows
    const float* Bt1  = Bb + (long)cc1 * ldb;
    const bool fullMN = (m0 + 16 <= M) && (n0 + 32 <= N);

    v8f acc0 = {0.f,0.f,0.f,0.f,0.f,0.f,0.f,0.f};
    v8f acc1 = {0.f,0.f,0.f,0.f,0.f,0.f,0.f,0.f};
    int k0 = 0;

    if (fullMN) {
        // ---- interior tile, full-K iterations: pure vector loads
        for (; k0 + 32 <= K; k0 += 32) {
            v16h af, bf0, bf1;
            {
                const float* p = Arow + k0 + ((lane >> 4) << 4);
                put8f(af, 0, *(const float4*)(p),      *(const float4*)(p + 4));
                put8f(af, 8, *(const float4*)(p + 8), *(const float4*)(p + 12));
            }
            if (transB) {
                const float* p0 = Bt0 + k0 + ((lane >> 4) << 4);
                const float* p1 = Bt1 + k0 + ((lane >> 4) << 4);
                put8f(bf0, 0, *(const float4*)(p0),     *(const float4*)(p0 + 4));
                put8f(bf0, 8, *(const float4*)(p0 + 8), *(const float4*)(p0 + 12));
                put8f(bf1, 0, *(const float4*)(p1),     *(const float4*)(p1 + 4));
                put8f(bf1, 8, *(const float4*)(p1 + 8), *(const float4*)(p1 + 12));
            } else {
                const int kb = k0 + ((lane >> 4) << 4);
                const float* p = Bb + (long)kb * ldb + col0;
#pragma unroll
                for (int j = 0; j < 16; ++j) {
                    bf0[j] = (half_t)p[(long)j * ldb];
                    bf1[j] = (half_t)p[(long)j * ldb + 16];
                }
            }
            acc0 = __builtin_amdgcn_wmma_f32_16x16x32_f16(false, af, false, bf0,
                                                          (short)0, acc0, false, false);
            acc1 = __builtin_amdgcn_wmma_f32_16x16x32_f16(false, af, false, bf1,
                                                          (short)0, acc1, false, false);
        }
        // ---- K tail: branchless clamped loads
        if (k0 < K) {
            v16h af, bf0, bf1;
#pragma unroll
            for (int j = 0; j < 16; ++j) {
                int k  = k0 + a_k_of(lane, j);
                int kc = (k < K) ? k : (K - 1);
                float v = Arow[kc];
                af[j] = (k < K) ? (half_t)v : (half_t)0.f;
            }
#pragma unroll
            for (int j = 0; j < 16; ++j) {
                int k  = k0 + b_k_of(lane, j);
                int kc = (k < K) ? k : (K - 1);
                float v0 = transB ? Bt0[kc] : Bb[(long)kc * ldb + col0];
                float v1 = transB ? Bt1[kc] : Bb[(long)kc * ldb + col1];
                bf0[j] = (k < K) ? (half_t)v0 : (half_t)0.f;
                bf1[j] = (k < K) ? (half_t)v1 : (half_t)0.f;
            }
            acc0 = __builtin_amdgcn_wmma_f32_16x16x32_f16(false, af, false, bf0,
                                                          (short)0, acc0, false, false);
            acc1 = __builtin_amdgcn_wmma_f32_16x16x32_f16(false, af, false, bf1,
                                                          (short)0, acc1, false, false);
        }
        float bn0 = 0.f, bn1 = 0.f;
        if (bias1) { bn0 += bias1[batch * sBias + col0]; bn1 += bias1[batch * sBias + col1]; }
        if (bias2) { bn0 += bias2[batch * sBias + col0]; bn1 += bias2[batch * sBias + col1]; }
#pragma unroll
        for (int vv = 0; vv < 8; ++vv) {
            int m = m0 + vv + ((lane >> 4) << 3);
            float x0 = acc0[vv] + bn0;
            float x1 = acc1[vv] + bn1;
            if (act == 1) { x0 = fmaxf(x0, 0.f); x1 = fmaxf(x1, 0.f); }
            Cb[(long)m * ldc + col0] = x0;
            Cb[(long)m * ldc + col1] = x1;
        }
    } else {
        // ---- edge tile: branchless clamped loads + selects everywhere
        for (k0 = 0; k0 < K; k0 += 32) {
            v16h af, bf0, bf1;
#pragma unroll
            for (int j = 0; j < 16; ++j) {
                int k  = k0 + a_k_of(lane, j);
                int kc = (k < K) ? k : (K - 1);
                float v = Arow[kc];
                af[j] = (rowOK && k < K) ? (half_t)v : (half_t)0.f;
            }
#pragma unroll
            for (int j = 0; j < 16; ++j) {
                int k  = k0 + b_k_of(lane, j);
                int kc = (k < K) ? k : (K - 1);
                float v0 = transB ? Bt0[kc] : Bb[(long)kc * ldb + cc0];
                float v1 = transB ? Bt1[kc] : Bb[(long)kc * ldb + cc1];
                bf0[j] = (col0OK && k < K) ? (half_t)v0 : (half_t)0.f;
                bf1[j] = (col1OK && k < K) ? (half_t)v1 : (half_t)0.f;
            }
            acc0 = __builtin_amdgcn_wmma_f32_16x16x32_f16(false, af, false, bf0,
                                                          (short)0, acc0, false, false);
            acc1 = __builtin_amdgcn_wmma_f32_16x16x32_f16(false, af, false, bf1,
                                                          (short)0, acc1, false, false);
        }
        float bn0 = 0.f, bn1 = 0.f;
        if (col0OK) {
            if (bias1) bn0 += bias1[batch * sBias + col0];
            if (bias2) bn0 += bias2[batch * sBias + col0];
        }
        if (col1OK) {
            if (bias1) bn1 += bias1[batch * sBias + col1];
            if (bias2) bn1 += bias2[batch * sBias + col1];
        }
#pragma unroll
        for (int vv = 0; vv < 8; ++vv) {
            int m = m0 + vv + ((lane >> 4) << 3);
            if (m < M) {
                float x0 = acc0[vv] + bn0;
                float x1 = acc1[vv] + bn1;
                if (act == 1) { x0 = fmaxf(x0, 0.f); x1 = fmaxf(x1, 0.f); }
                if (col0OK) Cb[(long)m * ldc + col0] = x0;
                if (col1OK) Cb[(long)m * ldc + col1] = x1;
            }
        }
    }
}

// ---------------------------------------------------------------------------
// Fused attention per (16-row tile, head, batch). qkv: [B][200][384] (q|k|v).
// S -> LDS padded to 224 cols: cols 200..207 = -1e30 (softmax kill),
// cols 208..223 = 0 (P@V k-loop needs no A-side masking). out: [B][200][128].
// ---------------------------------------------------------------------------
__global__ void attention_kernel(const float* __restrict__ qkv, float* __restrict__ out)
{
    const int NN = 200;
    const int m0   = blockIdx.x * 16;
    const int head = blockIdx.y;
    const int b    = blockIdx.z;

    __shared__ float s_S[16][224];
    __shared__ float s_red[16][17];
    __shared__ float s_stat[16];

    const int tid = threadIdx.x, lane = tid & 31, wave = tid >> 5;
    const float* qb = qkv + (long)b * 200 * 384;
    const float scale = 0.17677669529663687f; // 1/sqrt(32)

    // zero the far pad (cols 208..223), one element per thread
    s_S[tid >> 4][208 + (tid & 15)] = 0.f;

    // ---- phase 1: S = (Q @ K^T) * scale   (Dh=32 -> one WMMA k-step)
    {
        const int  rowm  = m0 + (lane & 15);
        const bool rowOK = rowm < NN;
        const int  rc    = rowOK ? rowm : (NN - 1);
        v16h af;
        {
            const float* p = qb + (long)rc * 384 + head * 32 + ((lane >> 4) << 4);
            float4 a0 = *(const float4*)(p);
            float4 a1 = *(const float4*)(p + 4);
            float4 a2 = *(const float4*)(p + 8);
            float4 a3 = *(const float4*)(p + 12);
            if (!rowOK) { a0 = {0,0,0,0}; a1 = a0; a2 = a0; a3 = a0; } // lane-level select
            put8f(af, 0, a0, a1);
            put8f(af, 8, a2, a3);
        }
        for (int ct = wave; ct < 13; ct += 8) {
            const int  coln  = ct * 16 + (lane & 15);
            const bool colOK = coln < NN;
            const int  cc    = colOK ? coln : (NN - 1);
            v16h bf;
            {
                const float* p = qb + (long)cc * 384 + 128 + head * 32 + ((lane >> 4) << 4);
                float4 b0 = *(const float4*)(p);
                float4 b1 = *(const float4*)(p + 4);
                float4 b2 = *(const float4*)(p + 8);
                float4 b3 = *(const float4*)(p + 12);
                if (!colOK) { b0 = {0,0,0,0}; b1 = b0; b2 = b0; b3 = b0; }
                put8f(bf, 0, b0, b1);
                put8f(bf, 8, b2, b3);
            }
            v8f acc = {0.f,0.f,0.f,0.f,0.f,0.f,0.f,0.f};
            acc = __builtin_amdgcn_wmma_f32_16x16x32_f16(false, af, false, bf,
                                                         (short)0, acc, false, false);
#pragma unroll
            for (int vv = 0; vv < 8; ++vv) {
                int m = vv + ((lane >> 4) << 3);
                s_S[m][coln] = colOK ? acc[vv] * scale : -1e30f;
            }
        }
    }
    __syncthreads();

    // ---- phase 2: row softmax over cols [0,208) (16 rows x 16 threads each)
    {
        int r = tid >> 4, c0 = tid & 15;
        float pm = -1e30f;
        for (int c = c0; c < 208; c += 16) pm = fmaxf(pm, s_S[r][c]);
        s_red[r][c0] = pm;
        __syncthreads();
        if (c0 == 0) { float mx = -1e30f; for (int i = 0; i < 16; ++i) mx = fmaxf(mx, s_red[r][i]); s_stat[r] = mx; }
        __syncthreads();
        float mx = s_stat[r], ps = 0.f;
        for (int c = c0; c < 208; c += 16) { float e = __expf(s_S[r][c] - mx); s_S[r][c] = e; ps += e; }
        s_red[r][c0] = ps;
        __syncthreads();
        if (c0 == 0) { float s = 0.f; for (int i = 0; i < 16; ++i) s += s_red[r][i]; s_stat[r] = 1.f / s; }
        __syncthreads();
        float inv = s_stat[r];
        for (int c = c0; c < 208; c += 16) s_S[r][c] *= inv;
    }
    __syncthreads();

    // ---- phase 3: O = P @ V (16x32, waves 0/1 take one 16-col tile each)
    if (wave < 2) {
        const int n0 = wave * 16;
        v8f acc = {0.f,0.f,0.f,0.f,0.f,0.f,0.f,0.f};
        for (int k0 = 0; k0 < 224; k0 += 32) {
            v16h af, bf;
            {   // P rows from LDS: pad cols are exact zeros, no masking needed
                const float* p = &s_S[lane & 15][k0 + ((lane >> 4) << 4)];
                put8f(af, 0, *(const float4*)(p),      *(const float4*)(p + 4));
                put8f(af, 8, *(const float4*)(p + 8), *(const float4*)(p + 12));
            }
#pragma unroll
            for (int j = 0; j < 16; ++j) {   // V: strided over nodes, clamp+select
                int k  = k0 + b_k_of(lane, j);
                int kc = (k < NN) ? k : (NN - 1);
                float v = qb[(long)kc * 384 + 256 + head * 32 + n0 + (lane & 15)];
                bf[j] = (k < NN) ? (half_t)v : (half_t)0.f;
            }
            acc = __builtin_amdgcn_wmma_f32_16x16x32_f16(false, af, false, bf,
                                                         (short)0, acc, false, false);
        }
#pragma unroll
        for (int vv = 0; vv < 8; ++vv) {
            int m = m0 + vv + ((lane >> 4) << 3);
            if (m < NN)
                out[(long)b * 200 * 128 + (long)m * 128 + head * 32 + n0 + (lane & 15)] = acc[vv];
        }
    }
}

// mean over 200 nodes: [B][200][128] -> [B][128]
__global__ void node_mean_kernel(const float* __restrict__ o, float* __restrict__ obar)
{
    int b = blockIdx.x, j = threadIdx.x; // 128 threads
    float s = 0.f;
    const float* p = o + (long)b * 200 * 128 + j;
    for (int n = 0; n < 200; ++n) s += p[n * 128];
    obar[b * 128 + j] = s * 0.005f; // 1/200
}

// combined[bt] = [ temporal(feats@temp_w^T + temp_b) | gfeat[b] ]  (256 wide)
__global__ void build_combined_kernel(
    const float* __restrict__ pos, const float* __restrict__ vel,
    const float* __restrict__ st,  const float* __restrict__ ac,
    const float* __restrict__ obj, const float* __restrict__ tl,
    const float* __restrict__ temp_w, const float* __restrict__ temp_b,
    const float* __restrict__ gfeat, float* __restrict__ combined)
{
    int bt = blockIdx.x;        // b*50 + t
    int b  = bt / 50;
    int tid = threadIdx.x;
    if (tid < 128) {
        float f[8];
        f[0] = pos[bt * 2]; f[1] = pos[bt * 2 + 1];
        f[2] = vel[bt * 2]; f[3] = vel[bt * 2 + 1];
        f[4] = st[bt]; f[5] = ac[bt]; f[6] = obj[bt]; f[7] = tl[bt];
        float s = temp_b[tid];
#pragma unroll
        for (int k = 0; k < 8; ++k) s += f[k] * temp_w[tid * 8 + k];
        combined[(long)bt * 256 + tid] = s;
    } else {
        int j = tid - 128;
        combined[(long)bt * 256 + 128 + j] = gfeat[b * 128 + j];
    }
}

// ---------------------------------------------------------------------------
// LSTM scan: per (16 batch rows, lstm). gates: [6][B*T][512] precomputed
// (x@wih^T + bih + bhh). whh staged to LDS as f16 (128KB). Each step:
// G = gates[t] + h @ whh^T (WMMA, accumulator preloaded from gates), then
// elementwise gate math. A-fragments (h) hoisted into registers once per
// step and reused across all 4 N-tiles (4x fewer ds_loads on the critical
// sequential path). LDS fragment reads are aligned 16/32B vector loads.
// ---------------------------------------------------------------------------
__global__ void lstm_scan_kernel(
    const float* __restrict__ gates, const float* __restrict__ whh,
    float* __restrict__ hseq, float* __restrict__ lasth,
    int T, int writeSeq)
{
    const int l  = blockIdx.z;
    const int m0 = blockIdx.x * 16;       // batch-row base
    extern __shared__ float smem[];
    half_t* s_whh = (half_t*)smem;                    // 512*128 halves
    float*  s_G   = (float*)(s_whh + 512 * 128);      // 16*512
    float*  s_c   = s_G + 16 * 512;                   // 16*128
    half_t* s_h   = (half_t*)(s_c + 16 * 128);        // 16*128 halves

    const int tid = threadIdx.x, lane = tid & 31, wave = tid >> 5;
    for (int i = tid; i < 512 * 128; i += 256) s_whh[i] = (half_t)whh[(long)l * 512 * 128 + i];
    for (int i = tid; i < 16 * 128; i += 256) { s_c[i] = 0.f; s_h[i] = (half_t)0.f; }
    __syncthreads();

    const float* gl = gates + (long)l * 12800 * 512;

    for (int t = 0; t < T; ++t) {
        // hoist the four A-fragments (rows of h) for this step
        v16h af4[4];
#pragma unroll
        for (int kk = 0; kk < 4; ++kk) {
            const h8* pa = (const h8*)(s_h + (lane & 15) * 128 + kk * 32 + ((lane >> 4) << 4));
            put8h(af4[kk], 0, pa[0]);
            put8h(af4[kk], 8, pa[1]);     // +8 halves
        }
        // recurrent GEMM: 16 x 512, K=128; wave handles 4 N-tiles
        for (int i = 0; i < 4; ++i) {
            const int n0   = (wave * 4 + i) * 16;
            const int coln = n0 + (lane & 15);
            v8f acc;
#pragma unroll
            for (int vv = 0; vv < 8; ++vv) {
                int m = vv + ((lane >> 4) << 3);
                acc[vv] = gl[((long)(m0 + m) * T + t) * 512 + coln];
            }
#pragma unroll
            for (int kk = 0; kk < 4; ++kk) {
                v16h bf;
                const h8* pb = (const h8*)(s_whh + (long)coln * 128 + kk * 32 + ((lane >> 4) << 4));
                put8h(bf, 0, pb[0]);
                put8h(bf, 8, pb[1]);
                acc = __builtin_amdgcn_wmma_f32_16x16x32_f16(false, af4[kk], false, bf,
                                                             (short)0, acc, false, false);
            }
#pragma unroll
            for (int vv = 0; vv < 8; ++vv) {
                int m = vv + ((lane >> 4) << 3);
                s_G[m * 512 + coln] = acc[vv];
            }
        }
        __syncthreads();

        // gate nonlinearity: i | f | g | o column blocks of 128
        for (int idx = tid; idx < 16 * 128; idx += 256) {
            int m = idx >> 7, j = idx & 127;
            float gi = s_G[m * 512 + j];
            float gf = s_G[m * 512 + 128 + j];
            float gg = s_G[m * 512 + 256 + j];
            float go = s_G[m * 512 + 384 + j];
            float c = sigf(gf) * s_c[idx] + sigf(gi) * tanhf(gg);
            float h = sigf(go) * tanhf(c);
            s_c[idx] = c;
            s_h[idx] = (half_t)h;
            if (writeSeq)
                hseq[(long)l * 12800 * 128 + ((long)(m0 + m) * T + t) * 128 + j] = h;
            else if (t == T - 1)
                lasth[(long)l * 256 * 128 + (m0 + m) * 128 + j] = h;
        }
        __syncthreads();
    }
}

// ---------------------------------------------------------------------------
// Output heads: 420 dot-128 per batch element, routed into the 10 flat outputs
// ---------------------------------------------------------------------------
__global__ void heads_kernel(
    const float* __restrict__ lasth,
    const float* __restrict__ pos_ow, const float* __restrict__ pos_ob,
    const float* __restrict__ vel_ow, const float* __restrict__ vel_ob,
    const float* __restrict__ st_ow,  const float* __restrict__ st_ob,
    const float* __restrict__ ac_ow,  const float* __restrict__ ac_ob,
    const float* __restrict__ obj_ow, const float* __restrict__ obj_ob,
    const float* __restrict__ tl_ow,  const float* __restrict__ tl_ob,
    float* __restrict__ out)
{
    const int b = blockIdx.x;
    for (int j = threadIdx.x; j < 420; j += 256) {
        int lstm, row; const float* W; const float* Bv;
        if      (j < 120) { lstm = 0; row = j;       W = pos_ow; Bv = pos_ob; }
        else if (j < 240) { lstm = 1; row = j - 120; W = vel_ow; Bv = vel_ob; }
        else if (j < 300) { lstm = 2; row = j - 240; W = st_ow;  Bv = st_ob;  }
        else if (j < 360) { lstm = 3; row = j - 300; W = ac_ow;  Bv = ac_ob;  }
        else if (j < 390) { lstm = 4; row = j - 360; W = obj_ow; Bv = obj_ob; }
        else              { lstm = 5; row = j - 390; W = tl_ow;  Bv = tl_ob;  }
        const float* h = lasth + ((long)lstm * 256 + b) * 128;
        float s = Bv[row];
        for (int k = 0; k < 128; ++k) s += h[k] * W[row * 128 + k];

        if (j < 120) {            // pos: (B,30,4) -> mean | softplus
            int to = row >> 2, c = row & 3;
            if (c < 2) out[        ((long)b * 30 + to) * 2 + c      ] = s;
            else       out[15360 + ((long)b * 30 + to) * 2 + (c - 2)] = softplusf(s);
        } else if (j < 240) {     // vel
            int to = row >> 2, c = row & 3;
            if (c < 2) out[30720 + ((long)b * 30 + to) * 2 + c      ] = s;
            else       out[46080 + ((long)b * 30 + to) * 2 + (c - 2)] = softplusf(s);
        } else if (j < 300) {     // steering
            int to = row >> 1, c = row & 1;
            if (c == 0) out[61440 + (long)b * 30 + to] = s;
            else        out[69120 + (long)b * 30 + to] = softplusf(s);
        } else if (j < 360) {     // acceleration
            int to = row >> 1, c = row & 1;
            if (c == 0) out[76800 + (long)b * 30 + to] = s;
            else        out[84480 + (long)b * 30 + to] = softplusf(s);
        } else if (j < 390) {     // object_in_path
            out[92160 + (long)b * 30 + row] = sigf(s);
        } else {                  // traffic_light
            out[99840 + (long)b * 30 + row] = sigf(s);
        }
    }
}

// ---------------------------------------------------------------------------
extern "C" void kernel_launch(void* const* d_in, const int* in_sizes, int n_in,
                              void* d_out, int out_size, void* d_ws, size_t ws_size,
                              hipStream_t stream)
{
    (void)in_sizes; (void)n_in; (void)out_size; (void)ws_size;
    const float* position   = (const float*)d_in[0];
    const float* velocity   = (const float*)d_in[1];
    const float* steering   = (const float*)d_in[2];
    const float* accel      = (const float*)d_in[3];
    const float* objp       = (const float*)d_in[4];
    const float* tlight     = (const float*)d_in[5];
    const float* node_feat  = (const float*)d_in[6];
    const float* adj        = (const float*)d_in[7];
    const float* gc1_w      = (const float*)d_in[8];
    const float* gc2_w      = (const float*)d_in[9];
    const float* attn_in_w  = (const float*)d_in[10];
    const float* attn_in_b  = (const float*)d_in[11];
    const float* attn_out_w = (const float*)d_in[12];
    const float* attn_out_b = (const float*)d_in[13];
    const float* temp_w     = (const float*)d_in[14];
    const float* temp_b     = (const float*)d_in[15];
    const float* wih0       = (const float*)d_in[16];
    const float* whh0       = (const float*)d_in[17];
    const float* bih0       = (const float*)d_in[18];
    const float* bhh0       = (const float*)d_in[19];
    const float* wih1       = (const float*)d_in[20];
    const float* whh1       = (const float*)d_in[21];
    const float* bih1       = (const float*)d_in[22];
    const float* bhh1       = (const float*)d_in[23];
    const float* pos_ow = (const float*)d_in[24]; const float* pos_ob = (const float*)d_in[25];
    const float* vel_ow = (const float*)d_in[26]; const float* vel_ob = (const float*)d_in[27];
    const float* st_ow  = (const float*)d_in[28]; const float* st_ob  = (const float*)d_in[29];
    const float* ac_ow  = (const float*)d_in[30]; const float* ac_ob  = (const float*)d_in[31];
    const float* obj_ow = (const float*)d_in[32]; const float* obj_ob = (const float*)d_in[33];
    const float* tl_ow  = (const float*)d_in[34]; const float* tl_ob  = (const float*)d_in[35];
    float* out = (float*)d_out;

    // workspace layout (floats)
    float* ws = (float*)d_ws;
    float* R0       = ws;                       // 51200*128  (XW / g1W / attn_o)
    float* R1       = R0 + 6553600;             // 51200*128  (g1 / g2)
    float* qkv      = R1 + 6553600;             // 51200*384
    float* obar     = qkv + 19660800;           // 256*128
    float* gfeat    = obar + 32768;             // 256*128
    float* combined = gfeat + 32768;            // 12800*256
    float* gatesb   = combined + 3276800;       // 6*12800*512 (reused L0 then L1)
    float* h1seq    = gatesb + 39321600;        // 6*12800*128
    float* lasth    = h1seq + 9830400;          // 6*256*128

    dim3 blk(256);

    // 1) XW = node_features(51200x16) @ gc1_w(16x128)
    wmma_gemm_kernel<<<dim3(800, 2, 1), blk, 0, stream>>>(
        node_feat, 16, 0, gc1_w, 128, 0, 0, nullptr, nullptr, 0,
        R0, 128, 0, 51200, 128, 16, 0);
    // 2) g1 = relu(adj @ XW)  batched over 256
    wmma_gemm_kernel<<<dim3(4, 2, 256), blk, 0, stream>>>(
        adj, 200, 40000, R0, 128, 25600, 0, nullptr, nullptr, 0,
        R1, 128, 25600, 200, 128, 200, 1);
    // 3) g1W = g1 @ gc2_w
    wmma_gemm_kernel<<<dim3(800, 2, 1), blk, 0, stream>>>(
        R1, 128, 0, gc2_w, 128, 0, 0, nullptr, nullptr, 0,
        R0, 128, 0, 51200, 128, 128, 0);
    // 4) g2 = relu(adj @ g1W)
    wmma_gemm_kernel<<<dim3(4, 2, 256), blk, 0, stream>>>(
        adj, 200, 40000, R0, 128, 25600, 0, nullptr, nullptr, 0,
        R1, 128, 25600, 200, 128, 200, 1);
    // 5) qkv = g2 @ attn_in_w^T + attn_in_b
    wmma_gemm_kernel<<<dim3(800, 6, 1), blk, 0, stream>>>(
        R1, 128, 0, attn_in_w, 128, 0, 1, attn_in_b, nullptr, 0,
        qkv, 384, 0, 51200, 384, 128, 0);
    // 6) attention -> R0
    attention_kernel<<<dim3(13, 4, 256), blk, 0, stream>>>(qkv, R0);
    // 7) mean over nodes
    node_mean_kernel<<<dim3(256), dim3(128), 0, stream>>>(R0, obar);
    // 8) gfeat = obar @ attn_out_w^T + b   (mean commutes with linear)
    wmma_gemm_kernel<<<dim3(4, 2, 1), blk, 0, stream>>>(
        obar, 128, 0, attn_out_w, 128, 0, 1, attn_out_b, nullptr, 0,
        gfeat, 128, 0, 256, 128, 128, 0);
    // 9) combined = [temporal | gfeat]
    build_combined_kernel<<<dim3(12800), blk, 0, stream>>>(
        position, velocity, steering, accel, objp, tlight, temp_w, temp_b,
        gfeat, combined);
    // 10) layer-0 input gates: combined @ wih0[l]^T + bih0[l] + bhh0[l]
    wmma_gemm_kernel<<<dim3(200, 8, 6), blk, 0, stream>>>(
        combined, 256, 0, wih0, 256, 131072, 1, bih0, bhh0, 512,
        gatesb, 512, 6553600, 12800, 512, 256, 0);
    // 11) layer-0 scan (whh in LDS as f16)
    size_t scan_lds = (size_t)512 * 128 * 2 + 16 * 512 * 4 + 16 * 128 * 4 + 16 * 128 * 2;
    lstm_scan_kernel<<<dim3(16, 1, 6), blk, scan_lds, stream>>>(
        gatesb, whh0, h1seq, lasth, 50, 1);
    // 12) layer-1 input gates: h1seq @ wih1[l]^T + bih1[l] + bhh1[l]
    wmma_gemm_kernel<<<dim3(200, 8, 6), blk, 0, stream>>>(
        h1seq, 128, 1638400, wih1, 128, 65536, 1, bih1, bhh1, 512,
        gatesb, 512, 6553600, 12800, 512, 128, 0);
    // 13) layer-1 scan, keep last h only
    lstm_scan_kernel<<<dim3(16, 1, 6), blk, scan_lds, stream>>>(
        gatesb, whh1, h1seq, lasth, 50, 0);
    // 14) output heads
    heads_kernel<<<dim3(256), blk, 0, stream>>>(
        lasth, pos_ow, pos_ob, vel_ow, vel_ob, st_ow, st_ob,
        ac_ow, ac_ob, obj_ow, obj_ob, tl_ow, tl_ob, out);
}
